// MoD_44719199486229
// MI455X (gfx1250) — matrix operand order repaired and checked by reading, compiled, not run on hardware
//
#include <hip/hip_runtime.h>
#include <stdint.h>

#define BB   2
#define SS   4096
#define DD   1024
#define HH   16
#define HDD  64
#define KSEL 2048
#define HID  4096
#define EPSF 1e-5f
#define SCALEF 0.125f

typedef __attribute__((ext_vector_type(16))) __bf16        v16bf;
typedef __attribute__((ext_vector_type(8)))  float         v8f;
typedef __attribute__((ext_vector_type(4)))  unsigned int  u32x4;
typedef __attribute__((ext_vector_type(4)))  int           v4i;

union Frag16 { v16bf v; unsigned short u[16]; u32x4 q[2]; };
union U16x8  { u32x4 q;    unsigned short u[8];  };
union U16x16 { u32x4 q[2]; unsigned short u[16]; };

// CDNA5 async global->LDS copy path (ASYNCcnt), guarded so either toolchain compiles.
#if defined(__HIP_DEVICE_COMPILE__) && \
    __has_builtin(__builtin_amdgcn_global_load_async_to_lds_b128) && \
    __has_builtin(__builtin_amdgcn_s_wait_asynccnt)
#define HAVE_ASYNC_LDS 1
#else
#define HAVE_ASYNC_LDS 0
#endif

#define ASG __attribute__((address_space(1)))
#define ASL __attribute__((address_space(3)))

__device__ __forceinline__ void lds_copy16(unsigned short* l, const unsigned short* g) {
#if HAVE_ASYNC_LDS
  __builtin_amdgcn_global_load_async_to_lds_b128((ASG v4i*)g, (ASL v4i*)l, 0, 0);
  __builtin_amdgcn_global_load_async_to_lds_b128((ASG v4i*)(g + 8), (ASL v4i*)(l + 8), 0, 0);
#else
  *(u32x4*)l       = *(const u32x4*)g;
  *(u32x4*)(l + 8) = *(const u32x4*)(g + 8);
#endif
}
__device__ __forceinline__ void async_wait() {
#if HAVE_ASYNC_LDS
  __builtin_amdgcn_s_wait_asynccnt(0);
#endif
}

__device__ __forceinline__ unsigned short f2bf(float f) {
  unsigned int u = __builtin_bit_cast(unsigned int, f);
  u += 0x7FFFu + ((u >> 16) & 1u);
  return (unsigned short)(u >> 16);
}
__device__ __forceinline__ float bf2f(unsigned short h) {
  unsigned int u = ((unsigned int)h) << 16;
  return __builtin_bit_cast(float, u);
}
__device__ __forceinline__ v8f wmma_bf16(v16bf a, v16bf b, v8f c) {
  return __builtin_amdgcn_wmma_f32_16x16x32_bf16(false, a, false, b,
                                                 (short)0, c, false, false);
}

// ---------------------------------------------------------------- utilities
__global__ __launch_bounds__(256)
void cvt_bf16_kernel(const float* __restrict__ s, unsigned short* __restrict__ d, int n) {
  int i = blockIdx.x * 256 + threadIdx.x;
  if (i < n) d[i] = f2bf(s[i]);
}

__global__ __launch_bounds__(256)
void copy4_kernel(const float4* __restrict__ s, float4* __restrict__ d, int n4) {
  int i = blockIdx.x * 256 + threadIdx.x;
  if (i < n4) d[i] = s[i];
}

// ---------------------------------------------------------------- router
__global__ __launch_bounds__(256)
void router_kernel(const float* __restrict__ x, const float* __restrict__ wr,
                   float* __restrict__ logits) {
  int lane = threadIdx.x & 31;
  int w    = threadIdx.x >> 5;
  int row  = blockIdx.x * 8 + w;
  const float* xr = x + (size_t)row * DD;
  float acc = 0.f;
#pragma unroll
  for (int i = 0; i < DD / 32; ++i) acc += xr[lane + 32 * i] * wr[lane + 32 * i];
#pragma unroll
  for (int m = 16; m >= 1; m >>= 1) acc += __shfl_xor(acc, m, 32);
  if (lane == 0) logits[row] = acc;
}

// ---------------------------------------------------------------- top-k
__global__ __launch_bounds__(1024)
void topk_kernel(const float* __restrict__ logits, int* __restrict__ sel,
                 float* __restrict__ rw) {
  __shared__ float sval[SS];
  __shared__ int   sidx[SS];
  __shared__ float red[1024];
  int tid = threadIdx.x;
  int b   = blockIdx.x;
  for (int i = tid; i < SS; i += 1024) { sval[i] = logits[b * SS + i]; sidx[i] = i; }
  __syncthreads();
  // bitonic: value desc, index asc tiebreak (matches jax top_k)
  for (int k = 2; k <= SS; k <<= 1) {
    for (int j = k >> 1; j > 0; j >>= 1) {
      for (int i = tid; i < SS; i += 1024) {
        int ixj = i ^ j;
        if (ixj > i) {
          float v1 = sval[i], v2 = sval[ixj];
          int   i1 = sidx[i], i2 = sidx[ixj];
          bool less = (v1 > v2) || (v1 == v2 && i1 < i2);
          bool up   = ((i & k) == 0);
          if (up ? !less : less) {
            sval[i] = v2; sval[ixj] = v1;
            sidx[i] = i2; sidx[ixj] = i1;
          }
        }
      }
      __syncthreads();
    }
  }
  // softmax over the selected K values
  float lm = -3.0e38f;
  for (int i = tid; i < KSEL; i += 1024) lm = fmaxf(lm, sval[i]);
  red[tid] = lm; __syncthreads();
  for (int st = 512; st > 0; st >>= 1) { if (tid < st) red[tid] = fmaxf(red[tid], red[tid + st]); __syncthreads(); }
  float mx = red[0]; __syncthreads();
  float ls = 0.f;
  for (int i = tid; i < KSEL; i += 1024) ls += __expf(sval[i] - mx);
  red[tid] = ls; __syncthreads();
  for (int st = 512; st > 0; st >>= 1) { if (tid < st) red[tid] += red[tid + st]; __syncthreads(); }
  float sm = red[0]; __syncthreads();
  // sort selected entries by token index ascending
  for (int k = 2; k <= KSEL; k <<= 1) {
    for (int j = k >> 1; j > 0; j >>= 1) {
      for (int i = tid; i < KSEL; i += 1024) {
        int ixj = i ^ j;
        if (ixj > i) {
          int i1 = sidx[i], i2 = sidx[ixj];
          bool less = i1 < i2;
          bool up   = ((i & k) == 0);
          if (up ? !less : less) {
            float v1 = sval[i]; sval[i] = sval[ixj]; sval[ixj] = v1;
            sidx[i] = i2; sidx[ixj] = i1;
          }
        }
      }
      __syncthreads();
    }
  }
  float inv = 1.f / sm;
  for (int i = tid; i < KSEL; i += 1024) {
    sel[b * KSEL + i] = sidx[i];
    rw[b * KSEL + i]  = __expf(sval[i] - mx) * inv;
  }
}

// ---------------------------------------------------------------- rmsnorm
__global__ __launch_bounds__(256)
void gather_rms_kernel(const float* __restrict__ x, const int* __restrict__ sel,
                       const float* __restrict__ wn, float* __restrict__ xf,
                       unsigned short* __restrict__ xnb) {
  __shared__ float red[256];
  int r   = blockIdx.x;
  int b   = r / KSEL;
  int src = sel[r];
  const float* xin = x + ((size_t)b * SS + src) * DD;
  float*       xo  = xf + (size_t)r * DD;
  int tid = threadIdx.x;
  float acc = 0.f;
  for (int i = tid; i < DD; i += 256) { float v = xin[i]; xo[i] = v; acc += v * v; }
  red[tid] = acc; __syncthreads();
  for (int st = 128; st > 0; st >>= 1) { if (tid < st) red[tid] += red[tid + st]; __syncthreads(); }
  float rms = rsqrtf(red[0] / (float)DD + EPSF);
  for (int i = tid; i < DD; i += 256)
    xnb[(size_t)r * DD + i] = f2bf(xo[i] * rms * wn[i]);
}

__global__ __launch_bounds__(256)
void rms_kernel(const float* __restrict__ in, const float* __restrict__ wn,
                unsigned short* __restrict__ outb) {
  __shared__ float red[256];
  int r = blockIdx.x;
  const float* xin = in + (size_t)r * DD;
  int tid = threadIdx.x;
  float acc = 0.f;
  for (int i = tid; i < DD; i += 256) { float v = xin[i]; acc += v * v; }
  red[tid] = acc; __syncthreads();
  for (int st = 128; st > 0; st >>= 1) { if (tid < st) red[tid] += red[tid + st]; __syncthreads(); }
  float rms = rsqrtf(red[0] / (float)DD + EPSF);
  for (int i = tid; i < DD; i += 256)
    outb[(size_t)r * DD + i] = f2bf(xin[i] * rms * wn[i]);
}

// ---------------------------------------------------------------- GEMM (bf16 WMMA, 128x128x32)
#define GBM 128
#define GBN 128
#define GBK 32
// epi: 0 = store f32, 1 = store bf16, 2 = f32 Add[]+acc, 3 = bf16 silu(Add[])*acc
__global__ __launch_bounds__(256)
void gemm_bf16_kernel(const unsigned short* __restrict__ A,
                      const unsigned short* __restrict__ Bm,
                      void* __restrict__ Cout,
                      const float* __restrict__ Add,
                      int M, int N, int Kd, int epi) {
  __shared__ unsigned short As[GBM * GBK];   // [m][k]
  __shared__ unsigned short Bs[GBN * GBK];   // [n][k] (transposed stage)
  int tid = threadIdx.x;
  int lane = tid & 31, w = tid >> 5;
  int wm = w >> 1, wn = w & 1;               // 4x2 wave grid; wave tile 32x64
  int row0 = blockIdx.y * GBM;
  int col0 = blockIdx.x * GBN;
  int l15  = lane & 15;
  int kb8  = (lane >> 4) * 8;   // A-fragment k base
  int kb16 = (lane >> 4) * 16;  // unused (B fragments also use kb8)

  v8f acc[2][4];
#pragma unroll
  for (int mt = 0; mt < 2; ++mt)
#pragma unroll
    for (int nt = 0; nt < 4; ++nt) {
      v8f z = {0.f,0.f,0.f,0.f,0.f,0.f,0.f,0.f};
      acc[mt][nt] = z;
    }

  int a_r = tid >> 1, a_c = (tid & 1) * 16;  // A stage: 16 ushorts/thread
  int b_k = tid >> 3, b_n = (tid & 7) * 16;  // B stage: 16 ushorts/thread

  for (int kk = 0; kk < Kd; kk += GBK) {
    // stage A tile (straight copy -> async-to-LDS when available)
    lds_copy16(As + a_r * GBK + a_c,
               A + (size_t)(row0 + a_r) * Kd + kk + a_c);
    // stage B tile transposed ([k][n] -> [n][k]) through VGPRs
    {
      U16x16 t;
      const unsigned short* bg = Bm + (size_t)(kk + b_k) * N + col0 + b_n;
      t.q[0] = *(const u32x4*)bg;
      t.q[1] = *(const u32x4*)(bg + 8);
#pragma unroll
      for (int e = 0; e < 16; ++e) Bs[(b_n + e) * GBK + b_k] = t.u[e];
    }
    if (kk + GBK < Kd) {
      __builtin_prefetch(A + (size_t)(row0 + a_r) * Kd + kk + GBK + a_c, 0, 1);
      __builtin_prefetch(Bm + (size_t)(kk + GBK + b_k) * N + col0 + b_n, 0, 1);
    }
    async_wait();
    __syncthreads();

    Frag16 af[2], bf;
#pragma unroll
    for (int mt = 0; mt < 2; ++mt) {
      int am = wm * 32 + mt * 16 + l15;
      af[mt].q[0] = *(const u32x4*)(As + am * GBK + kb8);
      af[mt].q[1] = *(const u32x4*)(As + am * GBK + kb8 + 16);
    }
#pragma unroll
    for (int nt = 0; nt < 4; ++nt) {
      int ni = wn * 64 + nt * 16 + l15;
      bf.q[0] = *(const u32x4*)(Bs + ni * GBK + kb8);
      bf.q[1] = *(const u32x4*)(Bs + ni * GBK + kb8 + 16);
      acc[0][nt] = wmma_bf16(af[0].v, bf.v, acc[0][nt]);
      acc[1][nt] = wmma_bf16(af[1].v, bf.v, acc[1][nt]);
    }
    __syncthreads();
  }

  int lhalf = (lane >> 4) << 3;
#pragma unroll
  for (int mt = 0; mt < 2; ++mt) {
#pragma unroll
    for (int nt = 0; nt < 4; ++nt) {
      int gc = col0 + wn * 64 + nt * 16 + l15;
#pragma unroll
      for (int j = 0; j < 8; ++j) {
        int gr = row0 + wm * 32 + mt * 16 + lhalf + j;
        size_t off = (size_t)gr * N + gc;
        float v = acc[mt][nt][j];
        if (epi == 0) {
          ((float*)Cout)[off] = v;
        } else if (epi == 1) {
          ((unsigned short*)Cout)[off] = f2bf(v);
        } else if (epi == 2) {
          ((float*)Cout)[off] = Add[off] + v;
        } else {
          float a0 = Add[off];
          float s0 = a0 / (1.f + __expf(-a0));
          ((unsigned short*)Cout)[off] = f2bf(s0 * v);
        }
      }
    }
  }
}

// ---------------------------------------------------------------- RoPE (in-place, bf16)
__global__ __launch_bounds__(256)
void rope_kernel(unsigned short* __restrict__ q, unsigned short* __restrict__ k,
                 const float* __restrict__ fc, const float* __restrict__ fs) {
  int p   = blockIdx.x * 256 + threadIdx.x;
  int d2  = p & (HDD / 2 - 1);
  int t   = p >> 5;
  int hh  = t & (HH - 1);
  int t2  = t >> 4;
  int pos = t2 & (KSEL - 1);
  int b   = t2 >> 11;
  float c = fc[pos * (HDD / 2) + d2];
  float s = fs[pos * (HDD / 2) + d2];
  size_t base = ((((size_t)b * KSEL + pos) * HH + hh) * HDD) + 2 * d2;
  float q1 = bf2f(q[base]), q2 = bf2f(q[base + 1]);
  q[base]     = f2bf(q1 * c - q2 * s);
  q[base + 1] = f2bf(q1 * s + q2 * c);
  float k1 = bf2f(k[base]), k2 = bf2f(k[base + 1]);
  k[base]     = f2bf(k1 * c - k2 * s);
  k[base + 1] = f2bf(k1 * s + k2 * c);
}

// ---------------------------------------------------------------- attention (flash, WMMA, 64-key steps)
__global__ __launch_bounds__(128)
void attn_kernel(const unsigned short* __restrict__ qb,
                 const unsigned short* __restrict__ kbuf,
                 const unsigned short* __restrict__ vbuf,
                 unsigned short* __restrict__ ob) {
  __shared__ unsigned short Kt[64 * 64];       // [key][d]
  __shared__ unsigned short Vt[64 * 64];       // [d][key]
  __shared__ unsigned short Ps[4 * 16 * 64];   // per-wave P stage [m][k]
  int tid  = threadIdx.x;
  int lane = tid & 31, w = tid >> 5;
  int l15  = lane & 15;
  int kb8  = (lane >> 4) * 8;
  int kb16 = (lane >> 4) * 16;
  int qtile = blockIdx.x, h = blockIdx.y, b = blockIdx.z;
  int qbase = qtile * 64;
  int qrow  = qbase + w * 16 + l15;
  size_t qoff = (((size_t)b * KSEL + qrow) * HH + h) * HDD;

  Frag16 qa0, qa1;
  qa0.q[0] = *(const u32x4*)(qb + qoff + kb8);
  qa0.q[1] = *(const u32x4*)(qb + qoff + kb8 + 16);
  qa1.q[0] = *(const u32x4*)(qb + qoff + 32 + kb8);
  qa1.q[1] = *(const u32x4*)(qb + qoff + 32 + kb8 + 16);

  float mrow[8], lrow[8];
  v8f oacc[4];
#pragma unroll
  for (int j = 0; j < 8; ++j) { mrow[j] = -3.0e38f; lrow[j] = 0.f; }
#pragma unroll
  for (int c = 0; c < 4; ++c) {
    v8f z = {0.f,0.f,0.f,0.f,0.f,0.f,0.f,0.f};
    oacc[c] = z;
  }

  int lv_key = tid >> 1;          // 0..63
  int lv_d   = (tid & 1) * 32;    // 0 or 32
  unsigned short* pw = Ps + w * (16 * 64);

  int nkb = qtile + 1;
  for (int kblk = 0; kblk < nkb; ++kblk) {
    int key0 = kblk * 64;
    __syncthreads();   // protect Kt/Vt reuse from previous iteration
    {
      size_t src = (((size_t)b * KSEL + key0 + lv_key) * HH + h) * HDD + lv_d;
      // K tile: straight copy (async path when available)
      lds_copy16(Kt + lv_key * 64 + lv_d,      kbuf + src);
      lds_copy16(Kt + lv_key * 64 + lv_d + 16, kbuf + src + 16);
      // V tile: transpose through VGPRs -> Vt[d][key]
      U16x16 vv;
      vv.q[0] = *(const u32x4*)(vbuf + src);
      vv.q[1] = *(const u32x4*)(vbuf + src + 8);
#pragma unroll
      for (int e = 0; e < 16; ++e) Vt[(lv_d + e) * 64 + lv_key] = vv.u[e];
      vv.q[0] = *(const u32x4*)(vbuf + src + 16);
      vv.q[1] = *(const u32x4*)(vbuf + src + 24);
#pragma unroll
      for (int e = 0; e < 16; ++e) Vt[(lv_d + 16 + e) * 64 + lv_key] = vv.u[e];
    }
    async_wait();
    __syncthreads();

    // S = Q K^T over a 16x64 score tile (4 key sub-tiles x 2 d-steps)
    v8f sfr[4];
#pragma unroll
    for (int t = 0; t < 4; ++t) {
      v8f z = {0.f,0.f,0.f,0.f,0.f,0.f,0.f,0.f};
      sfr[t] = z;
      Frag16 kf;
      int kn = t * 16 + l15;
      kf.q[0] = *(const u32x4*)(Kt + kn * 64 + kb8);
      kf.q[1] = *(const u32x4*)(Kt + kn * 64 + kb8 + 16);
      sfr[t] = wmma_bf16(qa0.v, kf.v, sfr[t]);
      kf.q[0] = *(const u32x4*)(Kt + kn * 64 + 32 + kb8);
      kf.q[1] = *(const u32x4*)(Kt + kn * 64 + 32 + kb8 + 16);
      sfr[t] = wmma_bf16(qa1.v, kf.v, sfr[t]);
    }

    float pv[4][8], alpha[8];
#pragma unroll
    for (int j = 0; j < 8; ++j) {
      int m  = j + ((lane >> 4) << 3);
      int qg = qbase + w * 16 + m;
      float t0 = -3.0e38f;
      float vals[4];
#pragma unroll
      for (int t = 0; t < 4; ++t) {
        int kg = key0 + t * 16 + l15;
        vals[t] = (kg <= qg) ? sfr[t][j] * SCALEF : -3.0e38f;
        t0 = fmaxf(t0, vals[t]);
      }
      t0 = fmaxf(t0, __shfl_xor(t0, 1, 32));
      t0 = fmaxf(t0, __shfl_xor(t0, 2, 32));
      t0 = fmaxf(t0, __shfl_xor(t0, 4, 32));
      t0 = fmaxf(t0, __shfl_xor(t0, 8, 32));
      float mnew = fmaxf(mrow[j], t0);
      float a = (mrow[j] > -1.0e38f) ? __expf(mrow[j] - mnew) : 0.f;
      float rs = 0.f;
#pragma unroll
      for (int t = 0; t < 4; ++t) {
        float e = __expf(vals[t] - mnew);
        pv[t][j] = e;
        rs += e;
      }
      rs += __shfl_xor(rs, 1, 32);
      rs += __shfl_xor(rs, 2, 32);
      rs += __shfl_xor(rs, 4, 32);
      rs += __shfl_xor(rs, 8, 32);
      lrow[j] = lrow[j] * a + rs;
      mrow[j] = mnew;
      alpha[j] = a;
    }
#pragma unroll
    for (int c = 0; c < 4; ++c)
#pragma unroll
      for (int j = 0; j < 8; ++j) oacc[c][j] *= alpha[j];
#pragma unroll
    for (int j = 0; j < 8; ++j) {
      int m = j + ((lane >> 4) << 3);
#pragma unroll
      for (int t = 0; t < 4; ++t)
        pw[m * 64 + t * 16 + l15] = f2bf(pv[t][j]);
    }
    __syncthreads();
    // O += P @ V  (two 32-key k-steps x 4 d-chunks)
    Frag16 pa0, pa1;
    pa0.q[0] = *(const u32x4*)(pw + l15 * 64 + kb8);
    pa0.q[1] = *(const u32x4*)(pw + l15 * 64 + kb8 + 16);
    pa1.q[0] = *(const u32x4*)(pw + l15 * 64 + 32 + kb8);
    pa1.q[1] = *(const u32x4*)(pw + l15 * 64 + 32 + kb8 + 16);
#pragma unroll
    for (int c = 0; c < 4; ++c) {
      Frag16 vf;
      int d = c * 16 + l15;
      vf.q[0] = *(const u32x4*)(Vt + d * 64 + kb8);
      vf.q[1] = *(const u32x4*)(Vt + d * 64 + kb8 + 16);
      oacc[c] = wmma_bf16(pa0.v, vf.v, oacc[c]);
      vf.q[0] = *(const u32x4*)(Vt + d * 64 + 32 + kb8);
      vf.q[1] = *(const u32x4*)(Vt + d * 64 + 32 + kb8 + 16);
      oacc[c] = wmma_bf16(pa1.v, vf.v, oacc[c]);
    }
  }

#pragma unroll
  for (int j = 0; j < 8; ++j) {
    float inv = (lrow[j] > 0.f) ? 1.f / lrow[j] : 0.f;
    int m  = j + ((lane >> 4) << 3);
    int qg = qbase + w * 16 + m;
    size_t oo = (((size_t)b * KSEL + qg) * HH + h) * HDD;
#pragma unroll
    for (int c = 0; c < 4; ++c)
      ob[oo + c * 16 + l15] = f2bf(oacc[c][j] * inv);
  }
}

// ---------------------------------------------------------------- scatter-add
__global__ __launch_bounds__(256)
void scatter_kernel(const float* __restrict__ y, const int* __restrict__ sel,
                    const float* __restrict__ rw, float* __restrict__ out) {
  int r = blockIdx.x;
  int b = r / KSEL;
  int s = sel[r];
  float wgt = rw[r];
  float*       dst = out + ((size_t)b * SS + s) * DD;
  const float* src = y + (size_t)r * DD;
  for (int i = threadIdx.x; i < DD; i += 256) dst[i] += wgt * src[i];
}

// ---------------------------------------------------------------- launch
extern "C" void kernel_launch(void* const* d_in, const int* in_sizes, int n_in,
                              void* d_out, int out_size, void* d_ws, size_t ws_size,
                              hipStream_t stream) {
  (void)in_sizes; (void)n_in; (void)out_size; (void)ws_size;
  const float* x         = (const float*)d_in[0];
  // d_in[1] = mask (causal tril) computed analytically in-kernel
  const float* fcos      = (const float*)d_in[2];
  const float* fsin      = (const float*)d_in[3];
  const float* wr        = (const float*)d_in[4];
  const float* attn_norm = (const float*)d_in[5];
  const float* wq        = (const float*)d_in[6];
  const float* wk        = (const float*)d_in[7];
  const float* wv        = (const float*)d_in[8];
  const float* wo        = (const float*)d_in[9];
  const float* ffn_norm  = (const float*)d_in[10];
  const float* w1        = (const float*)d_in[11];
  const float* w2        = (const float*)d_in[12];
  const float* w3        = (const float*)d_in[13];
  float* out = (float*)d_out;

  char* p = (char*)d_ws;
  auto carve = [&](size_t bytes) -> void* {
    void* r = (void*)p;
    p += (bytes + 255) & ~(size_t)255;
    return r;
  };
  const size_t MT = (size_t)BB * KSEL;  // 4096 compacted rows
  float*          logits = (float*)carve((size_t)BB * SS * 4);
  int*            sel    = (int*)carve(MT * 4);
  float*          rwt    = (float*)carve(MT * 4);
  float*          xf     = (float*)carve(MT * DD * 4);
  unsigned short* xnb    = (unsigned short*)carve(MT * DD * 2);
  unsigned short* qbb    = (unsigned short*)carve(MT * DD * 2);
  unsigned short* kbb    = (unsigned short*)carve(MT * DD * 2);
  unsigned short* vbb    = (unsigned short*)carve(MT * DD * 2);
  unsigned short* obb    = (unsigned short*)carve(MT * DD * 2);
  float*          hbuf   = (float*)carve(MT * DD * 4);
  unsigned short* hnb    = (unsigned short*)carve(MT * DD * 2);
  float*          a1     = (float*)carve(MT * (size_t)HID * 4);
  unsigned short* gbb    = (unsigned short*)carve(MT * (size_t)HID * 2);
  float*          ybuf   = (float*)carve(MT * DD * 4);
  unsigned short* wq_b   = (unsigned short*)carve((size_t)DD * DD * 2);
  unsigned short* wk_b   = (unsigned short*)carve((size_t)DD * DD * 2);
  unsigned short* wv_b   = (unsigned short*)carve((size_t)DD * DD * 2);
  unsigned short* wo_b   = (unsigned short*)carve((size_t)DD * DD * 2);
  unsigned short* w1_b   = (unsigned short*)carve((size_t)DD * HID * 2);
  unsigned short* w3_b   = (unsigned short*)carve((size_t)DD * HID * 2);
  unsigned short* w2_b   = (unsigned short*)carve((size_t)HID * DD * 2);

  const int nw  = DD * DD;
  const int nwh = DD * HID;
  cvt_bf16_kernel<<<(nw + 255) / 256, 256, 0, stream>>>(wq, wq_b, nw);
  cvt_bf16_kernel<<<(nw + 255) / 256, 256, 0, stream>>>(wk, wk_b, nw);
  cvt_bf16_kernel<<<(nw + 255) / 256, 256, 0, stream>>>(wv, wv_b, nw);
  cvt_bf16_kernel<<<(nw + 255) / 256, 256, 0, stream>>>(wo, wo_b, nw);
  cvt_bf16_kernel<<<(nwh + 255) / 256, 256, 0, stream>>>(w1, w1_b, nwh);
  cvt_bf16_kernel<<<(nwh + 255) / 256, 256, 0, stream>>>(w3, w3_b, nwh);
  cvt_bf16_kernel<<<(nwh + 255) / 256, 256, 0, stream>>>(w2, w2_b, nwh);

  router_kernel<<<(BB * SS) / 8, 256, 0, stream>>>(x, wr, logits);
  topk_kernel<<<BB, 1024, 0, stream>>>(logits, sel, rwt);
  gather_rms_kernel<<<(unsigned)MT, 256, 0, stream>>>(x, sel, attn_norm, xf, xnb);

  dim3 gq(DD / GBN, (unsigned)(MT / GBM));
  gemm_bf16_kernel<<<gq, 256, 0, stream>>>(xnb, wq_b, qbb, nullptr, (int)MT, DD, DD, 1);
  gemm_bf16_kernel<<<gq, 256, 0, stream>>>(xnb, wk_b, kbb, nullptr, (int)MT, DD, DD, 1);
  gemm_bf16_kernel<<<gq, 256, 0, stream>>>(xnb, wv_b, vbb, nullptr, (int)MT, DD, DD, 1);

  const int npairs = BB * KSEL * HH * (HDD / 2);
  rope_kernel<<<npairs / 256, 256, 0, stream>>>(qbb, kbb, fcos, fsin);

  attn_kernel<<<dim3(KSEL / 64, HH, BB), 128, 0, stream>>>(qbb, kbb, vbb, obb);

  gemm_bf16_kernel<<<gq, 256, 0, stream>>>(obb, wo_b, hbuf, xf, (int)MT, DD, DD, 2);
  rms_kernel<<<(unsigned)MT, 256, 0, stream>>>(hbuf, ffn_norm, hnb);

  dim3 gh(HID / GBN, (unsigned)(MT / GBM));
  gemm_bf16_kernel<<<gh, 256, 0, stream>>>(hnb, w1_b, a1, nullptr, (int)MT, HID, DD, 0);
  gemm_bf16_kernel<<<gh, 256, 0, stream>>>(hnb, w3_b, gbb, a1, (int)MT, HID, DD, 3);
  gemm_bf16_kernel<<<gq, 256, 0, stream>>>(gbb, w2_b, ybuf, hbuf, (int)MT, DD, HID, 2);

  const int n4 = BB * SS * DD / 4;
  copy4_kernel<<<(n4 + 255) / 256, 256, 0, stream>>>((const float4*)x, (float4*)out, n4);
  scatter_kernel<<<(unsigned)MT, 256, 0, stream>>>(ybuf, sel, rwt, out);
}
